// OptimizedAttention_13932873908457
// MI455X (gfx1250) — compile-verified, matrix operand order checked by reading, not run on hardware
//
#include <hip/hip_runtime.h>
#include <hip/hip_bf16.h>
#include <math.h>

// ---------------------------------------------------------------------------
// Problem constants (match reference)
// ---------------------------------------------------------------------------
constexpr int Bn   = 2;
constexpr int Sn   = 2048;
constexpr int Dn   = 2048;
constexpr int Hn   = 16;
constexpr int HKVn = 4;
constexpr int DHn  = 128;
constexpr float LOG_ROPE_BASE = 9.210340371976184f;   // ln(10000)
constexpr float INV_SQRT_DH   = 0.08838834764831845f; // 1/sqrt(128)

typedef __bf16 bf16;
typedef __attribute__((ext_vector_type(8)))  __bf16 bf16x8;
typedef __attribute__((ext_vector_type(16))) __bf16 bf16x16;
typedef __attribute__((ext_vector_type(8)))  float  floatx8;

static __device__ __forceinline__ bf16x16 make_frag(bf16x8 lo, bf16x8 hi) {
    bf16x16 r;
#pragma unroll
    for (int i = 0; i < 8; ++i) { r[i] = lo[i]; r[8 + i] = hi[i]; }
    return r;
}

static __device__ __forceinline__ floatx8 wmma_bf16(bf16x16 a, bf16x16 b, floatx8 c) {
    return __builtin_amdgcn_wmma_f32_16x16x32_bf16(false, a, false, b, (short)0, c,
                                                   false, false);
}

// ---------------------------------------------------------------------------
// Kernel 0a: elementwise f32 -> bf16 (for hidden states), 8 elems/thread
// ---------------------------------------------------------------------------
__global__ __launch_bounds__(256) void cvt_bf16_kernel(
    const float* __restrict__ src, bf16* __restrict__ dst)
{
    size_t i = ((size_t)blockIdx.x * 256 + threadIdx.x) * 8;
    const float4 a = *(const float4*)(src + i);
    const float4 b = *(const float4*)(src + i + 4);
    bf16x8 r;
    r[0] = (bf16)a.x; r[1] = (bf16)a.y; r[2] = (bf16)a.z; r[3] = (bf16)a.w;
    r[4] = (bf16)b.x; r[5] = (bf16)b.y; r[6] = (bf16)b.z; r[7] = (bf16)b.w;
    *(bf16x8*)(dst + i) = r;
}

// ---------------------------------------------------------------------------
// Kernel 0b: f32 [K,N] -> bf16 transposed [N,K] via 32x32 LDS tile
//   grid = (N/32, K/32), block = 256 (32x8)
// ---------------------------------------------------------------------------
__global__ __launch_bounds__(256) void cvt_transpose_kernel(
    const float* __restrict__ src, bf16* __restrict__ dst, int K, int N)
{
    __shared__ float t[32][33];
    const int tx = threadIdx.x & 31, ty = threadIdx.x >> 5;
    const int n0 = blockIdx.x * 32, k0 = blockIdx.y * 32;
#pragma unroll
    for (int i = 0; i < 4; ++i)
        t[ty + i * 8][tx] = src[(size_t)(k0 + ty + i * 8) * N + n0 + tx];
    __syncthreads();
#pragma unroll
    for (int i = 0; i < 4; ++i) {
        int n = ty + i * 8;
        dst[(size_t)(n0 + n) * K + k0 + tx] = (bf16)t[tx][n];
    }
}

// ---------------------------------------------------------------------------
// Kernel 1: fused QKV projection + RoPE.  Direct-from-L2 bf16 fragments,
// register-blocked 32 rows x 64 cols per wave (8 WMMA / 12 loads per K-step).
//   grid = (B*S/128, 24), block = 256 (8 waves; 4 row-groups x 2 col-groups)
//   Writes Qb[B,H,S,DH], Kb[B,HKV,S,DH], Vt[B,HKV,DH,S]  (bf16)
// ---------------------------------------------------------------------------
__global__ __launch_bounds__(256) void qkv_rope_kernel(
    const bf16* __restrict__ xb, const int* __restrict__ pos_ids,
    const bf16* __restrict__ wqT, const bf16* __restrict__ wkT,
    const bf16* __restrict__ wvT,
    bf16* __restrict__ Qb, bf16* __restrict__ Kb, bf16* __restrict__ Vt)
{
    __shared__ float sc[128 * 128];   // result tile for fused RoPE pairing (64KB)

    const int tid  = threadIdx.x;
    const int wave = tid >> 5;
    const int lane = tid & 31;
    const int hi   = lane >> 4;
    const int ln   = lane & 15;

    const int row0 = blockIdx.x * 128;
    const int ct   = blockIdx.y;

    const bf16* WT; int n0;
    if (ct < 16)      { WT = wqT; n0 = ct * 128; }
    else if (ct < 20) { WT = wkT; n0 = (ct - 16) * 128; }
    else              { WT = wvT; n0 = (ct - 20) * 128; }

    const int rbase = (wave & 3) * 32;      // 4 row-groups of 32
    const int cbase = (wave >> 2) * 64;     // 2 col-groups of 64

    const bf16* arow0 = xb + (size_t)(row0 + rbase + ln) * Dn;
    const bf16* arow1 = arow0 + (size_t)16 * Dn;
    const bf16* brow  = WT + (size_t)(n0 + cbase + ln) * Dn;

    floatx8 acc[2][4];
#pragma unroll
    for (int r = 0; r < 2; ++r)
#pragma unroll
        for (int j = 0; j < 4; ++j)
#pragma unroll
            for (int e = 0; e < 8; ++e) acc[r][j][e] = 0.0f;

#pragma unroll 2
    for (int k0 = 0; k0 < Dn; k0 += 32) {
        __builtin_prefetch(arow0 + k0 + 64, 0, 3);
        bf16x16 a0 = make_frag(*(const bf16x8*)(arow0 + k0 + hi * 8),
                               *(const bf16x8*)(arow0 + k0 + 16 + hi * 8));
        bf16x16 a1 = make_frag(*(const bf16x8*)(arow1 + k0 + hi * 8),
                               *(const bf16x8*)(arow1 + k0 + 16 + hi * 8));
#pragma unroll
        for (int j = 0; j < 4; ++j) {
            const bf16* br = brow + (size_t)j * 16 * Dn + k0;
            bf16x16 b = make_frag(*(const bf16x8*)(br + hi * 8),
                                  *(const bf16x8*)(br + 16 + hi * 8));
            acc[0][j] = wmma_bf16(a0, b, acc[0][j]);
            acc[1][j] = wmma_bf16(a1, b, acc[1][j]);
        }
    }

    // spill f32 results to LDS for cross-wave RoPE pairing (single barrier)
#pragma unroll
    for (int r = 0; r < 2; ++r)
#pragma unroll
        for (int j = 0; j < 4; ++j)
#pragma unroll
            for (int v = 0; v < 8; ++v)
                sc[(rbase + r * 16 + 8 * hi + v) * 128 + cbase + j * 16 + ln] =
                    acc[r][j][v];
    __syncthreads();

    const bool isQ = (ct < 16), isK = (ct >= 16 && ct < 20);
    const int head = isQ ? ct : (isK ? ct - 16 : ct - 20);

#pragma unroll 4
    for (int i = 0; i < 64; ++i) {
        int f = tid + i * 256;
        int r = f >> 7, d = f & 127;
        int g = row0 + r, b = g >> 11, s = g & (Sn - 1);
        float val = sc[r * 128 + d];
        if (isQ || isK) {
            float partner = sc[r * 128 + (d ^ 64)];
            float rot = (d < 64) ? -partner : partner;
            float pos = (float)pos_ids[b * Sn + s];
            float inv = __expf(-(float)(2 * (d & 63)) * (1.0f / DHn) * LOG_ROPE_BASE);
            float ang = pos * inv;
            val = val * __cosf(ang) + rot * __sinf(ang);
        }
        if (isQ)
            Qb[(((size_t)b * Hn + head) * Sn + s) * DHn + d] = (bf16)val;
        else if (isK)
            Kb[(((size_t)b * HKVn + head) * Sn + s) * DHn + d] = (bf16)val;
        else
            Vt[(((size_t)b * HKVn + head) * DHn + d) * Sn + s] = (bf16)val;
    }
}

// ---------------------------------------------------------------------------
// Kernel 2: causal GQA flash attention, bf16 WMMA, f32 online softmax
//   grid = (B*H, S/64), block = 128 (4 independent waves, 16 q-rows each)
// ---------------------------------------------------------------------------
__global__ __launch_bounds__(128) void attn_kernel(
    const bf16* __restrict__ Qb, const bf16* __restrict__ Kb,
    const bf16* __restrict__ Vt, bf16* __restrict__ attnO)
{
    __shared__ bf16 sp[4][16 * 32];   // per-wave P staging (C-layout -> A-layout)

    const int tid  = threadIdx.x;
    const int wave = tid >> 5;
    const int lane = tid & 31;
    const int hi   = lane >> 4;
    const int ln   = lane & 15;

    const int bh  = blockIdx.x;
    const int b   = bh / Hn, h = bh % Hn;
    const int hkv = h / (Hn / HKVn);
    const int q0  = blockIdx.y * 64 + wave * 16;

    const bf16* Qp = Qb + ((size_t)b * Hn + h) * Sn * DHn;
    const bf16* Kp = Kb + ((size_t)b * HKVn + hkv) * Sn * DHn;
    const bf16* Vp = Vt + ((size_t)b * HKVn + hkv) * DHn * Sn;

    // preload Q A-fragments (DH=128 -> 4 K-steps of 32)
    bf16x16 aq[4];
#pragma unroll
    for (int ds = 0; ds < 4; ++ds) {
        const bf16* qr = Qp + (size_t)(q0 + ln) * DHn + ds * 32;
        aq[ds] = make_frag(*(const bf16x8*)(qr + hi * 8),
                           *(const bf16x8*)(qr + 16 + hi * 8));
    }

    floatx8 oacc[8];
    float m[8], l[8];
#pragma unroll
    for (int j = 0; j < 8; ++j) {
#pragma unroll
        for (int v = 0; v < 8; ++v) oacc[j][v] = 0.0f;
        m[j] = -1e30f; l[j] = 0.0f;
    }

    bf16* pl = sp[wave];
    const int kend = q0 + 16;                     // causal bound

    for (int k0 = 0; k0 < kend; k0 += 32) {
        // ---- scores: S = Q K^T (two 16-col token subtiles) ----
        floatx8 s[2];
#pragma unroll
        for (int sub = 0; sub < 2; ++sub) {
#pragma unroll
            for (int v = 0; v < 8; ++v) s[sub][v] = 0.0f;
#pragma unroll
            for (int ds = 0; ds < 4; ++ds) {
                const bf16* kr =
                    Kp + (size_t)(k0 + sub * 16 + ln) * DHn + ds * 32;
                bf16x16 bk = make_frag(*(const bf16x8*)(kr + hi * 8),
                                       *(const bf16x8*)(kr + 16 + hi * 8));
                s[sub] = wmma_bf16(aq[ds], bk, s[sub]);
            }
        }
        // scale + causal mask
#pragma unroll
        for (int sub = 0; sub < 2; ++sub)
#pragma unroll
            for (int v = 0; v < 8; ++v) s[sub][v] *= INV_SQRT_DH;
        if (k0 + 31 > q0) {
#pragma unroll
            for (int sub = 0; sub < 2; ++sub)
#pragma unroll
                for (int v = 0; v < 8; ++v) {
                    int row = q0 + 8 * hi + v;
                    int col = k0 + sub * 16 + ln;
                    if (col > row) s[sub][v] = -1e30f;
                }
        }
        // ---- online softmax (row reductions across 16-lane N groups) ----
        float mt[8], rs[8], mn[8], scf[8];
#pragma unroll
        for (int v = 0; v < 8; ++v) {
            mt[v] = fmaxf(s[0][v], s[1][v]);
#pragma unroll
            for (int off = 1; off < 16; off <<= 1)
                mt[v] = fmaxf(mt[v], __shfl_xor(mt[v], off, 32));
            mn[v]  = fmaxf(m[v], mt[v]);
            scf[v] = __expf(m[v] - mn[v]);
            rs[v]  = 0.0f;
        }
#pragma unroll
        for (int sub = 0; sub < 2; ++sub)
#pragma unroll
            for (int v = 0; v < 8; ++v) {
                s[sub][v] = __expf(s[sub][v] - mn[v]);
                rs[v] += s[sub][v];
            }
#pragma unroll
        for (int v = 0; v < 8; ++v) {
#pragma unroll
            for (int off = 1; off < 16; off <<= 1)
                rs[v] += __shfl_xor(rs[v], off, 32);
            l[v] = l[v] * scf[v] + rs[v];
            m[v] = mn[v];
        }
#pragma unroll
        for (int j = 0; j < 8; ++j)
#pragma unroll
            for (int v = 0; v < 8; ++v) oacc[j][v] *= scf[v];

        // ---- P: C-layout -> A-layout via per-wave LDS patch ----
#pragma unroll
        for (int sub = 0; sub < 2; ++sub)
#pragma unroll
            for (int v = 0; v < 8; ++v)
                pl[(8 * hi + v) * 32 + sub * 16 + ln] = (bf16)s[sub][v];
        asm volatile("s_wait_dscnt 0" ::: "memory");
        bf16x16 ap = make_frag(*(const bf16x8*)&pl[ln * 32 + hi * 8],
                               *(const bf16x8*)&pl[ln * 32 + 16 + hi * 8]);

        // ---- O += P V  (V transposed: contiguous over tokens) ----
#pragma unroll
        for (int j = 0; j < 8; ++j) {
            const bf16* vr = Vp + (size_t)(j * 16 + ln) * Sn + k0;
            bf16x16 bv = make_frag(*(const bf16x8*)(vr + hi * 8),
                                   *(const bf16x8*)(vr + 16 + hi * 8));
            oacc[j] = wmma_bf16(ap, bv, oacc[j]);
        }
    }

    // normalize + store attn output [B, S, H*DH] bf16
#pragma unroll
    for (int v = 0; v < 8; ++v) l[v] = 1.0f / l[v];
#pragma unroll
    for (int j = 0; j < 8; ++j)
#pragma unroll
        for (int v = 0; v < 8; ++v) {
            int row = q0 + 8 * hi + v;
            int d   = j * 16 + ln;
            attnO[((size_t)b * Sn + row) * (Hn * DHn) + h * DHn + d] =
                (bf16)(oacc[j][v] * l[v]);
        }
}

// ---------------------------------------------------------------------------
// Kernel 3: output projection  out = attnO @ wo  (woT pre-transposed bf16)
//   Direct-global fragments, register-blocked 32x64 per wave, no LDS.
//   grid = (B*S/128, D/128), block = 256
// ---------------------------------------------------------------------------
__global__ __launch_bounds__(256) void oproj_kernel(
    const bf16* __restrict__ attnO, const bf16* __restrict__ woT,
    float* __restrict__ out)
{
    const int tid  = threadIdx.x;
    const int wave = tid >> 5;
    const int lane = tid & 31;
    const int hi   = lane >> 4;
    const int ln   = lane & 15;

    const int row0 = blockIdx.x * 128;
    const int col0 = blockIdx.y * 128;
    const int rbase = (wave & 3) * 32;
    const int cbase = (wave >> 2) * 64;

    const bf16* arow0 = attnO + (size_t)(row0 + rbase + ln) * Dn;
    const bf16* arow1 = arow0 + (size_t)16 * Dn;
    const bf16* brow  = woT + (size_t)(col0 + cbase + ln) * Dn;

    floatx8 acc[2][4];
#pragma unroll
    for (int r = 0; r < 2; ++r)
#pragma unroll
        for (int j = 0; j < 4; ++j)
#pragma unroll
            for (int e = 0; e < 8; ++e) acc[r][j][e] = 0.0f;

#pragma unroll 2
    for (int k0 = 0; k0 < Dn; k0 += 32) {
        __builtin_prefetch(arow0 + k0 + 64, 0, 3);
        bf16x16 a0 = make_frag(*(const bf16x8*)(arow0 + k0 + hi * 8),
                               *(const bf16x8*)(arow0 + k0 + 16 + hi * 8));
        bf16x16 a1 = make_frag(*(const bf16x8*)(arow1 + k0 + hi * 8),
                               *(const bf16x8*)(arow1 + k0 + 16 + hi * 8));
#pragma unroll
        for (int j = 0; j < 4; ++j) {
            const bf16* br = brow + (size_t)j * 16 * Dn + k0;
            bf16x16 b = make_frag(*(const bf16x8*)(br + hi * 8),
                                  *(const bf16x8*)(br + 16 + hi * 8));
            acc[0][j] = wmma_bf16(a0, b, acc[0][j]);
            acc[1][j] = wmma_bf16(a1, b, acc[1][j]);
        }
    }

#pragma unroll
    for (int r = 0; r < 2; ++r)
#pragma unroll
        for (int j = 0; j < 4; ++j)
#pragma unroll
            for (int v = 0; v < 8; ++v)
                out[(size_t)(row0 + rbase + r * 16 + 8 * hi + v) * Dn +
                    col0 + cbase + j * 16 + ln] = acc[r][j][v];
}

// ---------------------------------------------------------------------------
// Host-side launch
// ---------------------------------------------------------------------------
extern "C" void kernel_launch(void* const* d_in, const int* in_sizes, int n_in,
                              void* d_out, int out_size, void* d_ws, size_t ws_size,
                              hipStream_t stream) {
    (void)in_sizes; (void)n_in; (void)out_size; (void)ws_size;
    const float* x       = (const float*)d_in[0];
    const int*   pos_ids = (const int*)d_in[1];
    const float* wq      = (const float*)d_in[2];
    const float* wk      = (const float*)d_in[3];
    const float* wv      = (const float*)d_in[4];
    const float* wo      = (const float*)d_in[5];
    float* out = (float*)d_out;

    char* ws = (char*)d_ws;
    const size_t xBytes  = (size_t)Bn * Sn * Dn * sizeof(bf16);          // 16 MB
    const size_t wqBytes = (size_t)Dn * Hn * DHn * sizeof(bf16);         //  8 MB
    const size_t wkBytes = (size_t)Dn * HKVn * DHn * sizeof(bf16);       //  2 MB
    const size_t qBytes  = (size_t)Bn * Hn * Sn * DHn * sizeof(bf16);    // 16 MB
    const size_t kBytes  = (size_t)Bn * HKVn * Sn * DHn * sizeof(bf16);  //  4 MB

    size_t off = 0;
    bf16* xb    = (bf16*)(ws + off); off += xBytes;
    bf16* wqT   = (bf16*)(ws + off); off += wqBytes;
    bf16* wkT   = (bf16*)(ws + off); off += wkBytes;
    bf16* wvT   = (bf16*)(ws + off); off += wkBytes;
    bf16* woT   = (bf16*)(ws + off); off += wqBytes;
    bf16* Qb    = (bf16*)(ws + off); off += qBytes;
    bf16* Kb    = (bf16*)(ws + off); off += kBytes;
    bf16* Vt    = (bf16*)(ws + off); off += kBytes;
    bf16* attnO = xb;  // xb is dead after qkv_rope_kernel; reuse its region

    // 0) one-time bf16 conversion / weight transposition (L2-resident operands)
    cvt_bf16_kernel<<<dim3((Bn * Sn * Dn) / (256 * 8)), 256, 0, stream>>>(x, xb);
    cvt_transpose_kernel<<<dim3((Hn * DHn) / 32, Dn / 32), 256, 0, stream>>>(
        wq, wqT, Dn, Hn * DHn);
    cvt_transpose_kernel<<<dim3((HKVn * DHn) / 32, Dn / 32), 256, 0, stream>>>(
        wk, wkT, Dn, HKVn * DHn);
    cvt_transpose_kernel<<<dim3((HKVn * DHn) / 32, Dn / 32), 256, 0, stream>>>(
        wv, wvT, Dn, HKVn * DHn);
    cvt_transpose_kernel<<<dim3(Dn / 32, Dn / 32), 256, 0, stream>>>(
        wo, woT, Dn, Dn);

    // 1) fused QKV + RoPE   2) flash attention   3) output projection
    qkv_rope_kernel<<<dim3(Bn * Sn / 128, 24), 256, 0, stream>>>(
        xb, pos_ids, wqT, wkT, wvT, Qb, Kb, Vt);
    attn_kernel<<<dim3(Bn * Hn, Sn / 64), 128, 0, stream>>>(Qb, Kb, Vt, attnO);
    oproj_kernel<<<dim3(Bn * Sn / 128, Dn / 128), 256, 0, stream>>>(attnO, woT, out);
}
